// AutoF1LSTM_81990925681092
// MI455X (gfx1250) — compile-verified, matrix operand order checked by reading, not run on hardware
//
#include <hip/hip_runtime.h>

typedef __bf16 bf16_t;
typedef __attribute__((ext_vector_type(16))) __bf16 v16bf;
typedef __attribute__((ext_vector_type(8)))  float  v8f;

#define T_STEPS 256
#define BATCH   128
#define HID     1024
#define G3H     3072
#define KIN0    160     // 136 padded to multiple of 32
#define NHEAD_N 768     // 12 * 64
#define GRU_NB  96      // workgroups in persistent GRU kernel (2 col-tiles/wave)
#define GRU_NT  (GRU_NB * 256)

// ---------------- WMMA helpers ----------------

union FragU { v16bf v; uint4 q[2]; };

__device__ __forceinline__ v8f wmma_bf16(v16bf a, v16bf b, v8f c) {
  return __builtin_amdgcn_wmma_f32_16x16x32_bf16(false, a, false, b,
                                                 (short)0, c, false, false);
}

// A fragment from a hoisted per-lane pointer (already includes m*lda + k0).
__device__ __forceinline__ v16bf load_a_ptr(const bf16_t* p) {
  FragU f;
  f.q[0] = *(const uint4*)(p);
  f.q[1] = *(const uint4*)(p + 16);
  return f.v;
}

// B fragment: 32 contiguous bytes per lane (pre-packed layout), works for
// both global and LDS pointers (LDS -> ds_load_b128).
__device__ __forceinline__ v16bf load_b_ptr(const bf16_t* p) {
  FragU f;
  f.q[0] = *(const uint4*)(p);
  f.q[1] = *(const uint4*)(p + 8);
  return f.v;
}

// C/D 16x16 f32: VGPR v holds M=v (lanes<16) or M=8+v (lanes>=16), N=lane&15.
__device__ __forceinline__ void store_c(float* C, int ldc, v8f acc) {
  int lane = threadIdx.x & 31;
  int n  = lane & 15;
  int mb = (lane >> 4) << 3;
#pragma unroll
  for (int v = 0; v < 8; ++v)
    C[(size_t)(mb + v) * ldc + n] = acc[v];
}

__device__ __forceinline__ float sigmoidf_(float x) {
  return 1.0f / (1.0f + __expf(-x));
}

// ---------------- grid-wide barrier (persistent kernel) ----------------

__device__ __forceinline__ void grid_sync(unsigned* bar, unsigned nblocks) {
  __syncthreads();
  __threadfence();
  if (threadIdx.x == 0) {
    volatile unsigned* gen = bar + 1;
    unsigned g = *gen;
    if (atomicAdd(bar, 1u) == nblocks - 1) {
      *bar = 0;
      __threadfence();
      atomicAdd((unsigned*)gen, 1u);
    } else {
      while (*gen == g) { __builtin_amdgcn_s_sleep(2); }
    }
  }
  __syncthreads();
}

// ---------------- input assembly: lap + 9 embeddings -> bf16 (T*B,160) ----

__global__ void build_x_kernel(const float* __restrict__ lap,
                               const int* __restrict__ cat,
                               const float* __restrict__ e_team,
                               const float* __restrict__ e_track,
                               const float* __restrict__ e_drv,
                               const float* __restrict__ e_comp,
                               const float* __restrict__ e_stat,
                               bf16_t* __restrict__ xbf) {
  int row = blockIdx.x * blockDim.x + threadIdx.x;
  if (row >= T_STEPS * BATCH) return;
  bf16_t* dst = xbf + (size_t)row * KIN0;
  const float* lp = lap + (size_t)row * 64;
#pragma unroll 8
  for (int i = 0; i < 64; ++i) dst[i] = (bf16_t)lp[i];
  const int* ci = cat + (size_t)row * 9;
  const float* srcs[9] = {
    e_track + ci[4] * 8, e_track + ci[5] * 8, e_track + ci[6] * 8,
    e_track + ci[7] * 8, e_track + ci[3] * 8, e_team + ci[8] * 8,
    e_drv + ci[2] * 8, e_comp + ci[1] * 8, e_stat + ci[0] * 8 };
  int o = 64;
  for (int s = 0; s < 9; ++s)
#pragma unroll
    for (int i = 0; i < 8; ++i) dst[o++] = (bf16_t)srcs[s][i];
  for (; o < KIN0; ++o) dst[o] = (bf16_t)0.0f;
}

// ---------------- weight packing: W (N x Ksrc f32) -> WMMA-B fragments ----
// B[k][n] = W[n][k] (implicit transpose), zero-padded to Kp. Tile (kt,nt)
// occupies 512 consecutive bf16; a column strip (fixed nt) is contiguous.

__global__ void pack_b_kernel(const float* __restrict__ W, bf16_t* __restrict__ dst,
                              int N, int Ksrc, int Kp) {
  int g = blockIdx.x * blockDim.x + threadIdx.x;
  if (g >= N * Kp) return;
  int i    = g & 15;
  int lane = (g >> 4) & 31;
  int tile = g >> 9;
  int KT = Kp >> 5;
  int kt = tile % KT;
  int nt = tile / KT;
  int k0 = (lane >> 4) << 3;
  int k  = kt * 32 + k0 + ((i < 8) ? i : i + 8);
  int n  = nt * 16 + (lane & 15);
  float v = (k < Ksrc && n < N) ? W[(size_t)n * Ksrc + k] : 0.0f;
  dst[g] = (bf16_t)v;
}

// ---------------- persistent 2-phase GRU layer kernel ----------------
// grid = 96 x 256. WG owns column tiles (2*blockIdx, 2*blockIdx+1); the 8
// waves cover the 8 row tiles. B strips for both columns are staged in LDS
// ONCE (weights are step-invariant); A fragments are software-pipelined
// with the last K iteration peeled (no in-loop conditional -> no phi copies).

template<int KIN>
__global__ void __launch_bounds__(256)
gru_layer_kernel(const bf16_t* __restrict__ xin,
                 const bf16_t* __restrict__ WiP,
                 const bf16_t* __restrict__ WhP,
                 const float* __restrict__ bih,
                 const float* __restrict__ bhh,
                 const float* __restrict__ h_init,
                 float* __restrict__ h_f,
                 bf16_t* __restrict__ h_b,
                 float* __restrict__ gi_s,
                 float* __restrict__ gh_s,
                 bf16_t* __restrict__ y_bf,
                 float* __restrict__ hT_out,
                 unsigned* __restrict__ bar) {
  constexpr int KTI = KIN / 32;          // gi K-tiles (5 or 32)
  constexpr int KTH = HID / 32;          // gh K-tiles (32)
  constexpr int STRIP_I = KTI * 512;     // elements per gi column strip
  constexpr int STRIP_H = KTH * 512;     // elements per gh column strip
  extern __shared__ char smem_raw[];
  bf16_t* sm = (bf16_t*)smem_raw;
  // LDS layout: [Bi(tc0)][Bh(tc0)][Bi(tc1)][Bh(tc1)]
  bf16_t* smI0 = sm;
  bf16_t* smH0 = smI0 + STRIP_I;
  bf16_t* smI1 = smH0 + STRIP_H;
  bf16_t* smH1 = smI1 + STRIP_I;

  const unsigned NB = gridDim.x;
  const int tid  = blockIdx.x * 256 + threadIdx.x;
  const int warp = threadIdx.x >> 5;     // = tr (row tile)
  const int lane = threadIdx.x & 31;
  const int tc0  = blockIdx.x * 2;
  const int tc1  = tc0 + 1;

  // ---- stage B strips into LDS (once; weights constant over time) ----
  {
    const char* srcI0 = (const char*)(WiP + (size_t)tc0 * STRIP_I);
    const char* srcI1 = (const char*)(WiP + (size_t)tc1 * STRIP_I);
    const char* srcH0 = (const char*)(WhP + (size_t)tc0 * STRIP_H);
    const char* srcH1 = (const char*)(WhP + (size_t)tc1 * STRIP_H);
    for (int o = threadIdx.x * 16; o < STRIP_I * 2; o += 256 * 16) {
      *(uint4*)((char*)smI0 + o) = *(const uint4*)(srcI0 + o);
      *(uint4*)((char*)smI1 + o) = *(const uint4*)(srcI1 + o);
    }
    for (int o = threadIdx.x * 16; o < STRIP_H * 2; o += 256 * 16) {
      *(uint4*)((char*)smH0 + o) = *(const uint4*)(srcH0 + o);
      *(uint4*)((char*)smH1 + o) = *(const uint4*)(srcH1 + o);
    }
  }

  // ---- init hidden state from h_s ----
  for (int idx = tid; idx < BATCH * HID; idx += GRU_NT) {
    float v = h_init[idx];
    h_f[idx] = v;
    h_b[idx] = (bf16_t)v;
  }
  __syncthreads();
  grid_sync(bar, NB);

  // per-lane hoisted offsets (A fragment layout, ISA 7.12.2)
  const int m  = lane & 15;
  const int k0 = (lane >> 4) << 3;
  // hidden-state A pointer is invariant across timesteps
  const bf16_t* Ah_lane = h_b + (size_t)(warp * 16 + m) * HID + k0;
  const bf16_t* smI0_l = smI0 + lane * 16;
  const bf16_t* smH0_l = smH0 + lane * 16;
  const bf16_t* smI1_l = smI1 + lane * 16;
  const bf16_t* smH1_l = smH1 + lane * 16;

  float* gi_out0 = gi_s + (size_t)warp * 16 * G3H + tc0 * 16;
  float* gh_out0 = gh_s + (size_t)warp * 16 * G3H + tc0 * 16;

  for (int t = 0; t < T_STEPS; ++t) {
    // ---- phase 1: gi = x_t @ Wih^T ; gh = h @ Whh^T ----
    const bf16_t* Ax_lane = xin + ((size_t)t * BATCH + warp * 16 + m) * KIN + k0;

    // prefetch next timestep's x strip (global_prefetch_b8)
    if (t + 1 < T_STEPS)
      __builtin_prefetch(Ax_lane + (size_t)BATCH * KIN, 0, 1);

    // gi GEMM; A pipelined one K-tile ahead, last iteration peeled.
    // Stored immediately so only 2 accumulators stay live in the gh loop.
    {
      v8f aI0 = {0,0,0,0,0,0,0,0}, aI1 = {0,0,0,0,0,0,0,0};
      v16bf ax = load_a_ptr(Ax_lane);
#pragma unroll
      for (int kt = 0; kt < KTI - 1; ++kt) {
        v16bf axn = load_a_ptr(Ax_lane + (kt + 1) * 32);
        aI0 = wmma_bf16(ax, load_b_ptr(smI0_l + kt * 512), aI0);
        aI1 = wmma_bf16(ax, load_b_ptr(smI1_l + kt * 512), aI1);
        ax = axn;
      }
      aI0 = wmma_bf16(ax, load_b_ptr(smI0_l + (KTI - 1) * 512), aI0);
      aI1 = wmma_bf16(ax, load_b_ptr(smI1_l + (KTI - 1) * 512), aI1);
      store_c(gi_out0,      G3H, aI0);
      store_c(gi_out0 + 16, G3H, aI1);
    }

    // gh GEMM; same structure
    {
      v8f aH0 = {0,0,0,0,0,0,0,0}, aH1 = {0,0,0,0,0,0,0,0};
      v16bf ah = load_a_ptr(Ah_lane);
#pragma unroll 8
      for (int kt = 0; kt < KTH - 1; ++kt) {
        v16bf ahn = load_a_ptr(Ah_lane + (kt + 1) * 32);
        aH0 = wmma_bf16(ah, load_b_ptr(smH0_l + kt * 512), aH0);
        aH1 = wmma_bf16(ah, load_b_ptr(smH1_l + kt * 512), aH1);
        ah = ahn;
      }
      aH0 = wmma_bf16(ah, load_b_ptr(smH0_l + (KTH - 1) * 512), aH0);
      aH1 = wmma_bf16(ah, load_b_ptr(smH1_l + (KTH - 1) * 512), aH1);
      store_c(gh_out0,      G3H, aH0);
      store_c(gh_out0 + 16, G3H, aH1);
    }
    grid_sync(bar, NB);

    // ---- phase 2: gate math + hidden update ----
    for (int idx = tid; idx < BATCH * HID; idx += GRU_NT) {
      int b = idx >> 10;
      int j = idx & (HID - 1);
      const float* gib = gi_s + (size_t)b * G3H;
      const float* ghb = gh_s + (size_t)b * G3H;
      float r = sigmoidf_(gib[j] + bih[j] + ghb[j] + bhh[j]);
      float z = sigmoidf_(gib[HID + j] + bih[HID + j] + ghb[HID + j] + bhh[HID + j]);
      float n = tanhf(gib[2 * HID + j] + bih[2 * HID + j] +
                      r * (ghb[2 * HID + j] + bhh[2 * HID + j]));
      float hp = h_f[idx];
      float hn = (1.0f - z) * n + z * hp;
      h_f[idx] = hn;
      h_b[idx] = (bf16_t)hn;
      y_bf[(size_t)t * (BATCH * HID) + idx] = (bf16_t)hn;
      if (t == T_STEPS - 1) hT_out[idx] = hn;
    }
    grid_sync(bar, NB);
  }
}

// ---------------- LayerNorm (one wave32 per row of 1024) ----------------

__global__ void ln_kernel(const bf16_t* __restrict__ y,
                          const float* __restrict__ g,
                          const float* __restrict__ b,
                          bf16_t* __restrict__ out) {
  int row  = blockIdx.x * 8 + (threadIdx.x >> 5);
  int lane = threadIdx.x & 31;
  const bf16_t* yr = y + (size_t)row * HID;
  float vals[32];
  float s = 0.f, s2 = 0.f;
#pragma unroll
  for (int i = 0; i < 32; ++i) {
    float v = (float)yr[lane + i * 32];
    vals[i] = v; s += v; s2 += v * v;
  }
#pragma unroll
  for (int msk = 16; msk >= 1; msk >>= 1) {
    s  += __shfl_xor(s, msk);
    s2 += __shfl_xor(s2, msk);
  }
  float mean = s * (1.0f / HID);
  float var  = s2 * (1.0f / HID) - mean * mean;
  float rs = rsqrtf(var + 1e-5f);
  bf16_t* orow = out + (size_t)row * HID;
#pragma unroll
  for (int i = 0; i < 32; ++i) {
    int j = lane + i * 32;
    orow[j] = (bf16_t)((vals[i] - mean) * rs * g[j] + b[j]);
  }
}

// ---------------- 12-head first layer: (32768x1024)x(1024x768)+b, ReLU ----
// Each wave: one row tile x 2 column tiles; A and both B fragments are
// double-buffered (peeled final iteration), ~64 live VGPRs -> no spills.

__global__ void __launch_bounds__(256)
heads_gemm_kernel(const bf16_t* __restrict__ A,
                  const bf16_t* __restrict__ Bp,
                  const float* __restrict__ bias,
                  bf16_t* __restrict__ hid) {
  int waveId = blockIdx.x * 8 + (threadIdx.x >> 5);   // 0..49151
  int tr  = waveId / 24;                              // 0..2047
  int tcb = (waveId % 24) * 2;                        // 0,2,...,46
  int lane = threadIdx.x & 31;
  int m  = lane & 15;
  int k0 = (lane >> 4) << 3;
  constexpr int KT = HID / 32;
  const bf16_t* Al = A + (size_t)(tr * 16 + m) * HID + k0;
  const bf16_t* Bl0 = Bp + ((size_t)(tcb + 0) * KT << 9) + lane * 16;
  const bf16_t* Bl1 = Bp + ((size_t)(tcb + 1) * KT << 9) + lane * 16;

  v8f acc0 = {0,0,0,0,0,0,0,0}, acc1 = {0,0,0,0,0,0,0,0};

  v16bf a_c  = load_a_ptr(Al);
  v16bf b0_c = load_b_ptr(Bl0);
  v16bf b1_c = load_b_ptr(Bl1);

#pragma unroll 4
  for (int kt = 0; kt < KT - 1; ++kt) {
    v16bf a_n  = load_a_ptr(Al + (kt + 1) * 32);
    v16bf b0_n = load_b_ptr(Bl0 + (kt + 1) * 512);
    v16bf b1_n = load_b_ptr(Bl1 + (kt + 1) * 512);
    acc0 = wmma_bf16(a_c, b0_c, acc0);
    acc1 = wmma_bf16(a_c, b1_c, acc1);
    a_c = a_n; b0_c = b0_n; b1_c = b1_n;
  }
  acc0 = wmma_bf16(a_c, b0_c, acc0);
  acc1 = wmma_bf16(a_c, b1_c, acc1);

  int mb = (lane >> 4) << 3;
  {
    int n = (tcb + 0) * 16 + (lane & 15);
    float bb = bias[n];
#pragma unroll
    for (int v = 0; v < 8; ++v) {
      float x = acc0[v] + bb;
      x = x > 0.f ? x : 0.f;
      hid[(size_t)(tr * 16 + mb + v) * NHEAD_N + n] = (bf16_t)x;
    }
  }
  {
    int n = (tcb + 1) * 16 + (lane & 15);
    float bb = bias[n];
#pragma unroll
    for (int v = 0; v < 8; ++v) {
      float x = acc1[v] + bb;
      x = x > 0.f ? x : 0.f;
      hid[(size_t)(tr * 16 + mb + v) * NHEAD_N + n] = (bf16_t)x;
    }
  }
}

// ---------------- tiny second-layer heads into flattened d_out ----------

__global__ void heads_final_kernel(const bf16_t* __restrict__ hid,
                                   const float* __restrict__ cW, const float* __restrict__ cb,
                                   const float* __restrict__ gW, const float* __restrict__ gb,
                                   const float* __restrict__ sW, const float* __restrict__ sb,
                                   float* __restrict__ out) {
  int g = blockIdx.x * blockDim.x + threadIdx.x;
  const int ROWS = T_STEPS * BATCH;
  if (g >= ROWS * 24) return;
  int row = g / 24;
  int o   = g - row * 24;
  const bf16_t* hrow = hid + (size_t)row * NHEAD_N;
  const float* w;
  float bias;
  int head;
  size_t dst;
  if (o < 5) {                       // compound_decision (T,B,5)
    head = 0; w = cW + o * 64; bias = cb[o];
    dst = 262144 + (size_t)row * 5 + o;
  } else if (o < 14) {               // gear (T,B,9)
    int gg = o - 5;
    head = 8; w = gW + gg * 64; bias = gb[gg];
    dst = 655360 + (size_t)row * 9 + gg;
  } else {                           // 10 scalar heads
    int k = o - 14;
    const int hmap[10] = {1, 2, 3, 4, 5, 6, 7, 9, 10, 11};
    head = hmap[k]; w = sW + k * 64; bias = sb[k];
    if (k < 7)       dst = 425984 + (size_t)k * 32768 + row;   // lap..rpm
    else if (k == 7) dst = 950272 + (size_t)row;               // throttle
    else if (k == 8) dst = 983040 + (size_t)row;               // dist_ahead
    else             dst = 1015808 + (size_t)row;              // dist_behind
  }
  const bf16_t* h = hrow + head * 64;
  float acc = bias;
#pragma unroll 16
  for (int i = 0; i < 64; ++i) acc += (float)h[i] * w[i];
  out[dst] = acc;
}

// ---------------- host launch ----------------

extern "C" void kernel_launch(void* const* d_in, const int* in_sizes, int n_in,
                              void* d_out, int out_size, void* d_ws, size_t ws_size,
                              hipStream_t stream) {
  (void)in_sizes; (void)n_in; (void)out_size; (void)ws_size;
  const float* lap    = (const float*)d_in[0];
  const int*   cat    = (const int*)d_in[1];
  const float* h_s    = (const float*)d_in[2];
  const float* e_team = (const float*)d_in[3];
  const float* e_trk  = (const float*)d_in[4];
  const float* e_drv  = (const float*)d_in[5];
  const float* e_cmp  = (const float*)d_in[6];
  const float* e_sta  = (const float*)d_in[7];
  const float* Wih0   = (const float*)d_in[8];
  const float* Whh0   = (const float*)d_in[9];
  const float* bih0   = (const float*)d_in[10];
  const float* bhh0   = (const float*)d_in[11];
  const float* Wih1   = (const float*)d_in[12];
  const float* Whh1   = (const float*)d_in[13];
  const float* bih1   = (const float*)d_in[14];
  const float* bhh1   = (const float*)d_in[15];
  const float* ln_g   = (const float*)d_in[16];
  const float* ln_b   = (const float*)d_in[17];
  const float* headW1 = (const float*)d_in[18];
  const float* headb1 = (const float*)d_in[19];
  const float* cW2    = (const float*)d_in[20];
  const float* cb2    = (const float*)d_in[21];
  const float* gW2    = (const float*)d_in[22];
  const float* gb2    = (const float*)d_in[23];
  const float* sW2    = (const float*)d_in[24];
  const float* sb2    = (const float*)d_in[25];

  char* ws = (char*)d_ws;
  size_t off = 0;
  auto alloc = [&](size_t bytes) -> void* {
    void* p = ws + off;
    off = (off + bytes + 255) & ~(size_t)255;
    return p;
  };
  unsigned* bar  = (unsigned*)alloc(256);
  bf16_t* xbf  = (bf16_t*)alloc((size_t)32768 * KIN0 * 2);
  bf16_t* W0P  = (bf16_t*)alloc((size_t)G3H * KIN0 * 2);
  bf16_t* WH0P = (bf16_t*)alloc((size_t)G3H * HID * 2);
  bf16_t* W1P  = (bf16_t*)alloc((size_t)G3H * HID * 2);
  bf16_t* WH1P = (bf16_t*)alloc((size_t)G3H * HID * 2);
  bf16_t* HWP  = (bf16_t*)alloc((size_t)NHEAD_N * HID * 2);
  float*  gi_s = (float*)alloc((size_t)BATCH * G3H * 4);
  float*  gh_s = (float*)alloc((size_t)BATCH * G3H * 4);
  float*  h_f  = (float*)alloc((size_t)BATCH * HID * 4);
  bf16_t* h_b  = (bf16_t*)alloc((size_t)BATCH * HID * 2);
  bf16_t* y0   = (bf16_t*)alloc((size_t)32768 * HID * 2);
  bf16_t* y1   = (bf16_t*)alloc((size_t)32768 * HID * 2);
  bf16_t* ht   = (bf16_t*)alloc((size_t)32768 * HID * 2);
  bf16_t* hid  = (bf16_t*)alloc((size_t)32768 * NHEAD_N * 2);

  float* out = (float*)d_out;

  hipMemsetAsync(bar, 0, 256, stream);

  build_x_kernel<<<128, 256, 0, stream>>>(lap, cat, e_team, e_trk, e_drv,
                                          e_cmp, e_sta, xbf);

  pack_b_kernel<<<(G3H * KIN0 + 255) / 256, 256, 0, stream>>>(Wih0, W0P, G3H, 136, KIN0);
  pack_b_kernel<<<(G3H * HID) / 256, 256, 0, stream>>>(Whh0, WH0P, G3H, HID, HID);
  pack_b_kernel<<<(G3H * HID) / 256, 256, 0, stream>>>(Wih1, W1P, G3H, HID, HID);
  pack_b_kernel<<<(G3H * HID) / 256, 256, 0, stream>>>(Whh1, WH1P, G3H, HID, HID);
  pack_b_kernel<<<(NHEAD_N * HID) / 256, 256, 0, stream>>>(headW1, HWP, NHEAD_N, HID, HID);

  // dynamic LDS: 2 column strips of (Kin/32 + 32) KB each
  size_t lds0 = 2 * ((size_t)(KIN0 / 32) * 1024 + 32768);   //  75,776 B
  size_t lds1 = 2 * ((size_t)(HID  / 32) * 1024 + 32768);   // 131,072 B

  gru_layer_kernel<KIN0><<<GRU_NB, 256, lds0, stream>>>(
      xbf, W0P, WH0P, bih0, bhh0, h_s, h_f, h_b, gi_s, gh_s,
      y0, out, bar);
  gru_layer_kernel<HID><<<GRU_NB, 256, lds1, stream>>>(
      y0, W1P, WH1P, bih1, bhh1, h_s + (size_t)BATCH * HID,
      h_f, h_b, gi_s, gh_s, y1, out + 131072, bar);

  ln_kernel<<<32768 / 8, 256, 0, stream>>>(y1, ln_g, ln_b, ht);

  heads_gemm_kernel<<<(2048 * 24) / 8, 256, 0, stream>>>(ht, HWP, headb1, hid);

  heads_final_kernel<<<(32768 * 24 + 255) / 256, 256, 0, stream>>>(
      hid, cW2, cb2, gW2, gb2, sW2, sb2, out);
}
